// SelfAttention_RoPE_26929444946152
// MI455X (gfx1250) — matrix operand order repaired and checked by reading, not compiled, not run on hardware
//
#include <hip/hip_runtime.h>
#include <hip/hip_fp16.h>

// Problem constants
#define B_  8
#define L_  1024
#define C_  1024
#define H_  16
#define D_  64
#define MAXSCALE 4.6051701859880914f   // log(100)

typedef __attribute__((ext_vector_type(16))) _Float16 v16h;
typedef __attribute__((ext_vector_type(8)))  _Float16 v8h;
typedef __attribute__((ext_vector_type(4)))  _Float16 v4h;
typedef __attribute__((ext_vector_type(2)))  _Float16 v2h;
typedef __attribute__((ext_vector_type(8)))  float    v8f;

// ---- WMMA fragment helpers (layouts per CDNA5 ISA 7.12.2) ----------------
// A-matrix 16x32 f16: lane(0-15)=row M, K chunks {0..7,16..23}; lanes 16-31
// same rows, K chunks {8..15,24..31}. Source must be K-contiguous rows.
__device__ __forceinline__ v16h load_a_frag(const _Float16* p, int ld, int lane) {
  const int r = lane & 15, hf = lane >> 4;
  const _Float16* base = p + (size_t)r * ld + hf * 8;
  v8h lo = *(const v8h*)(base);
  v8h hi = *(const v8h*)(base + 16);
  v16h o;
#pragma unroll
  for (int i = 0; i < 8; ++i) { o[i] = lo[i]; o[i + 8] = hi[i]; }
  return o;
}

// B-matrix 32x16 f16: lane(0-15)=col N, K chunks {0..7,16..23}; lanes 16-31
// K chunks {8..15,24..31}. Source row n must be K-contiguous (i.e. "W^T" style storage).
__device__ __forceinline__ v16h load_b_frag(const _Float16* p, int ld, int lane) {
  const int n = lane & 15, hf = lane >> 4;
  const _Float16* base = p + (size_t)n * ld + hf * 8;
  v8h lo = *(const v8h*)(base);
  v8h hi = *(const v8h*)(base + 16);
  v16h o;
#pragma unroll
  for (int i = 0; i < 8; ++i) { o[i] = lo[i]; o[i + 8] = hi[i]; }
  return o;
}

__device__ __forceinline__ v8f wmma_f16(v16h a, v16h b, v8f c) {
  return __builtin_amdgcn_wmma_f32_16x16x32_f16(false, a, false, b, (short)0, c,
                                                false, false);
}

// ---- fp32 -> f16 convert --------------------------------------------------
__global__ __launch_bounds__(256) void cvt_f32_f16(const float* __restrict__ in,
                                                   _Float16* __restrict__ out,
                                                   int n4) {
  int i = blockIdx.x * 256 + threadIdx.x;
  if (i < n4) {
    float4 v = ((const float4*)in)[i];
    v4h o;
    o[0] = (_Float16)v.x; o[1] = (_Float16)v.y;
    o[2] = (_Float16)v.z; o[3] = (_Float16)v.w;
    ((v4h*)out)[i] = o;
  }
}

// ---- Fused QKV GEMM + bias + l2norm + scale + RoPE ------------------------
// Y[m,n] = sum_k X[m,k]*Wqkv[n,k] + bias[n]; M=8192, N=3072, K=1024.
// Wave computes 32(M)x64(N). Because nbase is a multiple of 64 and D=64,
// each wave's 64 columns are exactly one (s,h) block covering d=0..63, so the
// per-row l2-norm (reduction within 16-lane groups + 4 j-registers), the
// q-scale, and RoPE (pair partner = lane^1) run entirely in the epilogue.
// Writes q-hat / k-hat / v directly as f16 in [b][h][l][d].
__global__ __launch_bounds__(256) void qkv_gemm_kernel(
    const _Float16* __restrict__ X, const _Float16* __restrict__ W,
    const float* __restrict__ qbias, const float* __restrict__ vbias,
    const float* __restrict__ sml, const float* __restrict__ fcis,
    _Float16* __restrict__ QH, _Float16* __restrict__ KH,
    _Float16* __restrict__ VH) {
  const int lane = threadIdx.x & 31;
  const int wid  = blockIdx.x * 8 + (threadIdx.x >> 5);
  const int NT   = (3 * C_) / 64;                 // 48
  const int mbase = (wid / NT) * 32;
  const int nbase = (wid % NT) * 64;

  v8f z = {0.f,0.f,0.f,0.f,0.f,0.f,0.f,0.f};
  v8f acc[2][4];
#pragma unroll
  for (int i = 0; i < 2; ++i)
#pragma unroll
    for (int j = 0; j < 4; ++j) acc[i][j] = z;

  for (int kb = 0; kb < C_; kb += 32) {
    v16h a0 = load_a_frag(X + (size_t)mbase * C_ + kb, C_, lane);
    v16h a1 = load_a_frag(X + (size_t)(mbase + 16) * C_ + kb, C_, lane);
#pragma unroll
    for (int j = 0; j < 4; ++j) {
      v16h bf = load_b_frag(W + (size_t)(nbase + 16 * j) * C_ + kb, C_, lane);
      acc[0][j] = wmma_f16(a0, bf, acc[0][j]);
      acc[1][j] = wmma_f16(a1, bf, acc[1][j]);
    }
  }

  const int hf = lane >> 4, col = lane & 15;
  const int s = nbase >> 10;                 // 0=q, 1=k, 2=v (wave-uniform)
  const int h = (nbase & 1023) >> 6;
  float biasj[4];
#pragma unroll
  for (int j = 0; j < 4; ++j) {
    const int cidx = (nbase & 1023) + 16 * j + col;
    biasj[j] = (s == 0) ? qbias[cidx] : (s == 2 ? vbias[cidx] : 0.f);
  }

  if (s == 2) {
    // V: bias + f16, layout [b][h][l][d]
#pragma unroll
    for (int i = 0; i < 2; ++i) {
#pragma unroll
      for (int r = 0; r < 8; ++r) {
        const int m = mbase + 16 * i + r + 8 * hf;
        const int bb = m >> 10, l = m & 1023;
        _Float16* row = VH + (((size_t)bb * H_ + h) * L_ + l) * D_;
#pragma unroll
        for (int j = 0; j < 4; ++j)
          row[16 * j + col] = (_Float16)(acc[i][j][r] + biasj[j]);
      }
    }
  } else {
    // Q/K: bias, l2-norm over d=0..63, (q only) * exp(min(log_scale,log100)),
    // then RoPE with pair partner in lane^1, write f16 [b][h][l][d].
    const float scale = (s == 0) ? __expf(fminf(sml[h], MAXSCALE)) : 1.0f;
    _Float16* dst = (s == 0) ? QH : KH;
    const float2* fc2 = (const float2*)fcis;   // (cos, sin) pairs, [L][32]
#pragma unroll
    for (int i = 0; i < 2; ++i) {
#pragma unroll
      for (int r = 0; r < 8; ++r) {
        const int m = mbase + 16 * i + r + 8 * hf;
        const int bb = m >> 10, l = m & 1023;
        float vj[4];
        float ss = 0.f;
#pragma unroll
        for (int j = 0; j < 4; ++j) {
          vj[j] = acc[i][j][r] + biasj[j];
          ss += vj[j] * vj[j];
        }
        ss += __shfl_xor(ss, 1, 32);
        ss += __shfl_xor(ss, 2, 32);
        ss += __shfl_xor(ss, 4, 32);
        ss += __shfl_xor(ss, 8, 32);
        const float rn = scale / fmaxf(sqrtf(ss), 1e-12f);
        _Float16* row = dst + (((size_t)bb * H_ + h) * L_ + l) * D_;
#pragma unroll
        for (int j = 0; j < 4; ++j) {
          const float v = vj[j] * rn;
          const float other = __shfl_xor(v, 1, 32);     // rope pair partner
          const float2 cs = fc2[l * 32 + 8 * j + (col >> 1)];
          // even d: a*c - b*s ; odd d: a*s + b*c
          const float o = (col & 1) ? (other * cs.y + v * cs.x)
                                    : (v * cs.x - other * cs.y);
          row[16 * j + col] = (_Float16)o;
        }
      }
    }
  }
}

// ---- V transpose: f16 [b,h,l,d] -> f16 [b,h,d,l] (LDS staged) -------------
__global__ __launch_bounds__(256) void vtrans_kernel(const _Float16* __restrict__ VH,
                                                     _Float16* __restrict__ VT) {
  __shared__ _Float16 tile[64][65];
  const int bh = blockIdx.x >> 4;            // b*H+h
  const int lt = blockIdx.x & 15;            // L/64 tile
  const int t  = threadIdx.x;
  const _Float16* src = VH + ((size_t)bh * L_ + lt * 64) * D_;
#pragma unroll
  for (int i = 0; i < 4; ++i) {
    int idx = (i * 256 + t) * 4;             // 0..4095, 4-elem chunks
    int l = idx >> 6, d = idx & 63;
    v4h v = *(const v4h*)(src + idx);        // coalesced 8B reads
    tile[l][d + 0] = v[0]; tile[l][d + 1] = v[1];
    tile[l][d + 2] = v[2]; tile[l][d + 3] = v[3];
  }
  __syncthreads();
  _Float16* dst = VT + (size_t)bh * D_ * L_ + lt * 64;
#pragma unroll
  for (int i = 0; i < 4; ++i) {
    int idx = (i * 256 + t) * 4;
    int d = idx >> 6, l = idx & 63;
    v4h v;
    v[0] = tile[l + 0][d]; v[1] = tile[l + 1][d];
    v[2] = tile[l + 2][d]; v[3] = tile[l + 3][d];
    *(v4h*)(dst + (size_t)d * L_ + l) = v;   // coalesced 8B writes over l
  }
}

// ---- Flash attention: per-wave 16-query tile, online softmax --------------
__global__ __launch_bounds__(128) void attn_kernel(
    const _Float16* __restrict__ QH, const _Float16* __restrict__ KH,
    const _Float16* __restrict__ VT, const float* __restrict__ abias,
    _Float16* __restrict__ OH) {
  __shared__ __align__(16) _Float16 plds[4][16 * 32];
  const int lane = threadIdx.x & 31;
  const int warp = threadIdx.x >> 5;
  const int wid  = blockIdx.x * 4 + warp;    // 0..8191
  const int bh = wid >> 6;                   // 0..127
  const int qbase = (wid & 63) * 16;
  const int bidx = bh >> 4, h = bh & 15;
  const int hf = lane >> 4, col = lane & 15;

  const _Float16* qp = QH + ((size_t)bh * L_ + qbase) * D_;
  const _Float16* kp = KH + (size_t)bh * L_ * D_;
  const _Float16* vp = VT + (size_t)bh * D_ * L_;
  _Float16* my = &plds[warp][0];

  const v16h aq0 = load_a_frag(qp, D_, lane);
  const v16h aq1 = load_a_frag(qp + 32, D_, lane);

  v8f z = {0.f,0.f,0.f,0.f,0.f,0.f,0.f,0.f};
  v8f o0 = z, o1 = z, o2 = z, o3 = z;
  float mrow[8], lrow[8];
#pragma unroll
  for (int r = 0; r < 8; ++r) { mrow[r] = -3.0e38f; lrow[r] = 0.f; }

  for (int kb = 0; kb < L_; kb += 32) {
    // S tiles (16q x 16k) for the two key sub-tiles; D-layout col=k, row=q
    v8f s0 = z, s1 = z;
    {
      v16h b0 = load_b_frag(kp + (size_t)kb * D_, D_, lane);
      v16h b1 = load_b_frag(kp + (size_t)kb * D_ + 32, D_, lane);
      s0 = wmma_f16(aq0, b0, s0);
      s0 = wmma_f16(aq1, b1, s0);
      v16h b2 = load_b_frag(kp + (size_t)(kb + 16) * D_, D_, lane);
      v16h b3 = load_b_frag(kp + (size_t)(kb + 16) * D_ + 32, D_, lane);
      s1 = wmma_f16(aq0, b2, s1);
      s1 = wmma_f16(aq1, b3, s1);
    }
    // + attn_bias (broadcast over b,h), coalesced along k across lanes
#pragma unroll
    for (int r = 0; r < 8; ++r) {
      const size_t brow = (size_t)(qbase + r + 8 * hf) * L_ + kb + col;
      s0[r] += abias[brow];
      s1[r] += abias[brow + 16];
    }
    // online softmax (row reductions stay within 16-lane groups)
    float p0[8], p1[8], alpha[8];
#pragma unroll
    for (int r = 0; r < 8; ++r) {
      float v = fmaxf(s0[r], s1[r]);
      v = fmaxf(v, __shfl_xor(v, 1, 32));
      v = fmaxf(v, __shfl_xor(v, 2, 32));
      v = fmaxf(v, __shfl_xor(v, 4, 32));
      v = fmaxf(v, __shfl_xor(v, 8, 32));
      const float mn = fmaxf(mrow[r], v);
      alpha[r] = __expf(mrow[r] - mn);
      mrow[r] = mn;
      p0[r] = __expf(s0[r] - mn);
      p1[r] = __expf(s1[r] - mn);
      float rs = p0[r] + p1[r];
      rs += __shfl_xor(rs, 1, 32);
      rs += __shfl_xor(rs, 2, 32);
      rs += __shfl_xor(rs, 4, 32);
      rs += __shfl_xor(rs, 8, 32);
      lrow[r] = lrow[r] * alpha[r] + rs;
    }
#pragma unroll
    for (int r = 0; r < 8; ++r) {
      o0[r] *= alpha[r]; o1[r] *= alpha[r];
      o2[r] *= alpha[r]; o3[r] *= alpha[r];
    }
    // P: D-layout -> A-layout via per-wave LDS tile [q][k] (k contiguous)
#pragma unroll
    for (int r = 0; r < 8; ++r) {
      my[(r + 8 * hf) * 32 + col]      = (_Float16)p0[r];
      my[(r + 8 * hf) * 32 + col + 16] = (_Float16)p1[r];
    }
    asm volatile("s_wait_dscnt 0" ::: "memory");
    v16h pa = load_a_frag(my, 32, lane);
    // O += P(16x32) * V(32k x 64d); VT rows are k-contiguous
    v16h bv0 = load_b_frag(vp + (size_t)(0 * 16) * L_ + kb, L_, lane);
    v16h bv1 = load_b_frag(vp + (size_t)(1 * 16) * L_ + kb, L_, lane);
    v16h bv2 = load_b_frag(vp + (size_t)(2 * 16) * L_ + kb, L_, lane);
    v16h bv3 = load_b_frag(vp + (size_t)(3 * 16) * L_ + kb, L_, lane);
    o0 = wmma_f16(pa, bv0, o0);
    o1 = wmma_f16(pa, bv1, o1);
    o2 = wmma_f16(pa, bv2, o2);
    o3 = wmma_f16(pa, bv3, o3);
  }
  // epilogue: normalize and write f16 into [b][l][C] for the proj GEMM
#pragma unroll
  for (int r = 0; r < 8; ++r) {
    const float inv = 1.0f / lrow[r];
    const int l = qbase + r + 8 * hf;
    _Float16* row = OH + ((size_t)bidx * L_ + l) * C_ + h * 64;
    row[0 * 16 + col] = (_Float16)(o0[r] * inv);
    row[1 * 16 + col] = (_Float16)(o1[r] * inv);
    row[2 * 16 + col] = (_Float16)(o2[r] * inv);
    row[3 * 16 + col] = (_Float16)(o3[r] * inv);
  }
}

// ---- Output projection: out[m,n] = sum_k A[m,k]*Wp[n,k] + bproj[n] --------
__global__ __launch_bounds__(256) void proj_gemm_kernel(
    const _Float16* __restrict__ A, const _Float16* __restrict__ W,
    const float* __restrict__ bias, float* __restrict__ out) {
  const int lane = threadIdx.x & 31;
  const int wid  = blockIdx.x * 8 + (threadIdx.x >> 5);
  const int NT   = C_ / 64;                       // 16
  const int mbase = (wid / NT) * 32;
  const int nbase = (wid % NT) * 64;

  v8f z = {0.f,0.f,0.f,0.f,0.f,0.f,0.f,0.f};
  v8f acc[2][4];
#pragma unroll
  for (int i = 0; i < 2; ++i)
#pragma unroll
    for (int j = 0; j < 4; ++j) acc[i][j] = z;

  for (int kb = 0; kb < C_; kb += 32) {
    v16h a0 = load_a_frag(A + (size_t)mbase * C_ + kb, C_, lane);
    v16h a1 = load_a_frag(A + (size_t)(mbase + 16) * C_ + kb, C_, lane);
#pragma unroll
    for (int j = 0; j < 4; ++j) {
      v16h bf = load_b_frag(W + (size_t)(nbase + 16 * j) * C_ + kb, C_, lane);
      acc[0][j] = wmma_f16(a0, bf, acc[0][j]);
      acc[1][j] = wmma_f16(a1, bf, acc[1][j]);
    }
  }
  const int hf = lane >> 4, col = lane & 15;
#pragma unroll
  for (int i = 0; i < 2; ++i) {
#pragma unroll
    for (int j = 0; j < 4; ++j) {
      const int n = nbase + 16 * j + col;
      const float bn = bias[n];
#pragma unroll
      for (int r = 0; r < 8; ++r) {
        const int m = mbase + 16 * i + r + 8 * hf;
        out[(size_t)m * C_ + n] = acc[i][j][r] + bn;
      }
    }
  }
}

extern "C" void kernel_launch(void* const* d_in, const int* in_sizes, int n_in,
                              void* d_out, int out_size, void* d_ws, size_t ws_size,
                              hipStream_t stream) {
  (void)in_sizes; (void)n_in; (void)out_size; (void)ws_size;
  const float* x     = (const float*)d_in[0];
  const float* fcis  = (const float*)d_in[1];
  const float* abias = (const float*)d_in[2];
  const float* wqkv  = (const float*)d_in[3];
  const float* qbias = (const float*)d_in[4];
  const float* vbias = (const float*)d_in[5];
  const float* sml   = (const float*)d_in[6];
  const float* wproj = (const float*)d_in[7];
  const float* bproj = (const float*)d_in[8];
  float* out = (float*)d_out;

  char* ws = (char*)d_ws;
  size_t cur = 0;
  auto alloc = [&](size_t bytes) -> void* {
    void* p = ws + cur;
    cur += (bytes + 255) & ~(size_t)255;
    return p;
  };
  const size_t NTOK = (size_t)B_ * L_;            // 8192
  const size_t NQK  = (size_t)B_ * H_ * L_ * D_;  // 8388608
  _Float16* XH  = (_Float16*)alloc(NTOK * C_ * 2);
  _Float16* WQH = (_Float16*)alloc((size_t)3 * C_ * C_ * 2);
  _Float16* WPH = (_Float16*)alloc((size_t)C_ * C_ * 2);
  _Float16* QH  = (_Float16*)alloc(NQK * 2);
  _Float16* KH  = (_Float16*)alloc(NQK * 2);
  _Float16* VH  = (_Float16*)alloc(NQK * 2);
  _Float16* VT  = (_Float16*)alloc(NQK * 2);
  _Float16* OH  = (_Float16*)alloc(NTOK * C_ * 2);

  // 1) converts (x, W_qkv, W_proj)
  cvt_f32_f16<<<(int)(NTOK * C_ / 4 / 256), 256, 0, stream>>>(x, XH, (int)(NTOK * C_ / 4));
  cvt_f32_f16<<<3 * C_ * C_ / 4 / 256, 256, 0, stream>>>(wqkv, WQH, 3 * C_ * C_ / 4);
  cvt_f32_f16<<<C_ * C_ / 4 / 256, 256, 0, stream>>>(wproj, WPH, C_ * C_ / 4);
  // 2) fused QKV GEMM + bias + norm + scale + RoPE:
  //    (8192/32)*(3072/64)=12288 waves / 8 per block
  qkv_gemm_kernel<<<1536, 256, 0, stream>>>(XH, WQH, qbias, vbias, sml, fcis,
                                            QH, KH, VH);
  // 3) V transpose: (B*H)*(L/64)=2048 blocks
  vtrans_kernel<<<2048, 256, 0, stream>>>(VH, VT);
  // 4) flash attention: B*H*(L/16)=8192 waves, 4 waves/block
  attn_kernel<<<2048, 128, 0, stream>>>(QH, KH, VT, abias, OH);
  // 5) output projection: (8192/32)*(1024/64)=4096 waves / 8 per block
  proj_gemm_kernel<<<512, 256, 0, stream>>>(OH, WPH, bproj, out);
}
